// redu_dim_51539608236
// MI455X (gfx1250) — matrix operand order repaired and checked by reading, not compiled, not run on hardware
//
#include <hip/hip_runtime.h>

// Shapes from the reference
#define BATCH 32
#define SEQ   512
#define DIM   512
#define NEXP  8
#define POUT  96
#define NHEAD 4
#define DK    128
#define NTOK  (BATCH * SEQ)   // 16384

typedef unsigned int u32;
typedef __attribute__((ext_vector_type(16))) __bf16 v16bf;
typedef __attribute__((ext_vector_type(8)))  float  v8f;
typedef __attribute__((ext_vector_type(4)))  u32    v4u;
typedef __attribute__((ext_vector_type(8)))  int    v8i;
typedef __attribute__((ext_vector_type(4)))  int    v4i;

union ABFrag { v16bf v; u32 u[8]; uint4 q[2]; };

static __device__ __forceinline__ unsigned short f2bf(float f) {
    u32 u = __float_as_uint(f);
    u += 0x7FFFu + ((u >> 16) & 1u);   // round-to-nearest-even
    return (unsigned short)(u >> 16);
}
static __device__ __forceinline__ u32 pack2bf(float a, float b) {
    return (u32)f2bf(a) | ((u32)f2bf(b) << 16);
}

// ---------------------------------------------------------------------------
// Tensor Data Mover path (gfx1250): guarded so the file still compiles if the
// builtin is absent; arity differs between ROCm7.2/clang-22 (5 args) and
// amdgpu-toolchain/clang-23 (6 args).
// ---------------------------------------------------------------------------
#if defined(__gfx1250__) && __has_builtin(__builtin_amdgcn_tensor_load_to_lds) && \
    __has_builtin(__builtin_amdgcn_s_wait_tensorcnt)
#define USE_TDM 1
#endif

#ifdef USE_TDM
// LDS byte offset of a __shared__ object (ptrtoint of the AS(3) pointer)
static __device__ __forceinline__ u32 lds_off(void* p) {
    return (u32)(unsigned long long)(__attribute__((address_space(3))) char*)p;
}

// 2D bf16 tile load Global->LDS with hardware row padding:
// tile_d0 elems per row (32 bf16 = 16 DWORDs), pad +4 DWORDs per 16 DWORDs
// => LDS row stride 80B (matches LDSTR=20 below).
static __device__ __forceinline__ void tdm_load_2d_bf16(
    u32 lds_byte_off, const void* gaddr,
    u32 tensor_d0, u32 tensor_d1, u32 tile_d0, u32 tile_d1, u32 d0_stride)
{
    unsigned long long ga = (unsigned long long)gaddr;
    v4u g0;
    g0[0] = 1u;                                   // count=1 (valid user D#)
    g0[1] = lds_byte_off;                         // lds_addr
    g0[2] = (u32)ga;                              // global_addr[31:0]
    g0[3] = (u32)((ga >> 32) & 0x01FFFFFFull) | (2u << 30);  // addr[56:32] | type=2
    v8i g1;
    g1[0] = (int)((1u << 16)        // data_size = 2 bytes
                | (1u << 20)        // pad_enable
                | (3u << 22)        // pad_interval: 16 DWORDs
                | (3u << 25));      // pad_amount : 4 DWORDs
    g1[1] = (int)((tensor_d0 & 0xFFFFu) << 16);                       // dim0[15:0]
    g1[2] = (int)((tensor_d0 >> 16) | ((tensor_d1 & 0xFFFFu) << 16)); // dim0[31:16]|dim1[15:0]
    g1[3] = (int)((tensor_d1 >> 16) | (tile_d0 << 16));               // dim1[31:16]|tile0
    g1[4] = (int)(tile_d1 & 0xFFFFu);                                 // tile1 | tile2=0
    g1[5] = (int)d0_stride;                                           // stride0[31:0]
    g1[6] = 0;                                                        // stride0[47:32]|stride1 lo
    g1[7] = 0;
    v4i g2 = {0, 0, 0, 0};
    v4i g3 = {0, 0, 0, 0};
#if __clang_major__ >= 23
    v8i g4 = {0, 0, 0, 0, 0, 0, 0, 0};
    __builtin_amdgcn_tensor_load_to_lds(g0, g1, g2, g3, g4, 0);
#else
    __builtin_amdgcn_tensor_load_to_lds(g0, g1, g2, g3, 0);
#endif
}
#endif // USE_TDM

// ---------------------------------------------------------------------------
// Pre-pass: f32 -> packed bf16 (row-major, layout preserved)
// ---------------------------------------------------------------------------
__global__ __launch_bounds__(256) void cvt_bf16_kernel(
    const float* __restrict__ src, u32* __restrict__ dst, int n2)
{
    int g = blockIdx.x * 256 + threadIdx.x;
    if (g < n2) {
        const float2 f = ((const float2*)src)[g];
        dst[g] = pack2bf(f.x, f.y);
    }
}

// Pre-pass: W[K][N] f32 -> Wt[N][K] packed bf16 (K-major for GEMM B operand)
__global__ __launch_bounds__(256) void cvt_T_kernel(
    const float* __restrict__ W, u32* __restrict__ Wt, int K, int N)
{
    int g = blockIdx.x * 256 + threadIdx.x;   // over N * K/2, n fastest (coalesced reads)
    int n  = g & (N - 1);
    int kq = g >> 9;                          // N == 512
    if (kq < (K >> 1))
        Wt[(size_t)n * (K >> 1) + kq] =
            pack2bf(W[(size_t)(2 * kq) * N + n], W[(size_t)(2 * kq + 1) * N + n]);
}

// ---------------------------------------------------------------------------
// WMMA GEMM: C[M,N] = A[M,K] * Bt[N,K]^T + bias[N]
// A, Bt are packed bf16 (u32 pairs, K-major). Block tile 128x64, 8 waves
// (4x2), each wave 32x32 = 2x2 WMMA 16x16x32 tiles. TDM double-buffered
// staging when available; plain b128 copies otherwise.
// ---------------------------------------------------------------------------
#define BM 128
#define BN 64
#define BK 32
#define LDSTR 20   // uints per LDS row (16 data DWORDs + 4 pad) = 80B

__global__ __launch_bounds__(256) void gemm_bias_bf16wmma(
    const u32* __restrict__ A2, const u32* __restrict__ Bt2,
    const float* __restrict__ bias, float* __restrict__ C,
    int M, int N, int Kd)
{
    __shared__ __align__(16) u32 As[2][BM * LDSTR];
    __shared__ __align__(16) u32 Bs[2][BN * LDSTR];

    const int tid  = threadIdx.x;
    const int lane = tid & 31;
    const int wid  = tid >> 5;
    const int wm   = wid & 3;        // 4 waves along M
    const int wn   = wid >> 2;       // 2 waves along N
    const int m0   = blockIdx.y * BM;
    const int n0   = blockIdx.x * BN;
    const int K2   = Kd >> 1;        // u32 per row

    const int r  = lane & 15;
    const int hi = lane >> 4;

    v8f acc[2][2] = {};

#ifdef USE_TDM
    const u32 asOff[2] = { lds_off(&As[0][0]), lds_off(&As[1][0]) };
    const u32 bsOff[2] = { lds_off(&Bs[0][0]), lds_off(&Bs[1][0]) };
    if (wid == 0) {
        tdm_load_2d_bf16(asOff[0], A2  + (size_t)m0 * K2, Kd, M, BK, BM, Kd);
        tdm_load_2d_bf16(bsOff[0], Bt2 + (size_t)n0 * K2, Kd, N, BK, BN, Kd);
        __builtin_amdgcn_s_wait_tensorcnt(0);
    }
    __syncthreads();
#endif

    int buf = 0;
    for (int kk = 0; kk < Kd; kk += BK) {
#ifdef USE_TDM
        const bool has_next = (kk + BK) < Kd;
        if (wid == 0 && has_next) {   // async prefetch of next K-slice into other buffer
            tdm_load_2d_bf16(asOff[buf ^ 1], A2  + (size_t)m0 * K2 + ((kk + BK) >> 1),
                             Kd, M, BK, BM, Kd);
            tdm_load_2d_bf16(bsOff[buf ^ 1], Bt2 + (size_t)n0 * K2 + ((kk + BK) >> 1),
                             Kd, N, BK, BN, Kd);
        }
#else
        // synchronous staging: pure 16B copies (operands already bf16, K-major)
        #pragma unroll
        for (int c = 0; c < 2; ++c) {
            int chunk = tid + c * 256;          // 0..511
            int row = chunk >> 2, cq = chunk & 3;
            uint4 d = *(const uint4*)(A2 + (size_t)(m0 + row) * K2 + (kk >> 1) + cq * 4);
            *(uint4*)&As[0][row * LDSTR + cq * 4] = d;
        }
        {
            int row = tid >> 2, cq = tid & 3;   // 0..255
            uint4 d = *(const uint4*)(Bt2 + (size_t)(n0 + row) * K2 + (kk >> 1) + cq * 4);
            *(uint4*)&Bs[0][row * LDSTR + cq * 4] = d;
        }
        if (kk + BK < Kd)
            __builtin_prefetch(A2 + (size_t)(m0 + (tid >> 1)) * K2 + ((kk + BK) >> 1), 0, 1);
        __syncthreads();
#endif

        // ---- fragments + WMMA ----
        ABFrag a[2], b[2];
        #pragma unroll
        for (int mi = 0; mi < 2; ++mi) {
            const u32* ab = &As[buf][(wm * 32 + mi * 16 + r) * LDSTR];
            a[mi].q[0] = *(const uint4*)(ab + hi * 8);        // K 0-15 / 16-31
            a[mi].q[1] = *(const uint4*)(ab + hi * 8 + 4);
        }
        #pragma unroll
        for (int ni = 0; ni < 2; ++ni) {
            const u32* bb = &Bs[buf][(wn * 32 + ni * 16 + r) * LDSTR];
            b[ni].q[0] = *(const uint4*)(bb + hi * 8);        // K 0-15 / 16-31
            b[ni].q[1] = *(const uint4*)(bb + hi * 8 + 4);
        }
        #pragma unroll
        for (int mi = 0; mi < 2; ++mi)
            #pragma unroll
            for (int ni = 0; ni < 2; ++ni)
                acc[mi][ni] = __builtin_amdgcn_wmma_f32_16x16x32_bf16(
                    false, a[mi].v, false, b[ni].v,
                    (short)0, acc[mi][ni], false, false);
        __syncthreads();

#ifdef USE_TDM
        if (wid == 0 && has_next) __builtin_amdgcn_s_wait_tensorcnt(0);
        __syncthreads();
        buf ^= 1;
#endif
    }

    // ---- epilogue: C = acc + bias ----
    #pragma unroll
    for (int mi = 0; mi < 2; ++mi)
        #pragma unroll
        for (int ni = 0; ni < 2; ++ni) {
            int col = n0 + wn * 32 + ni * 16 + r;
            float bv = bias[col];
            #pragma unroll
            for (int q = 0; q < 8; ++q) {
                int row = m0 + wm * 32 + mi * 16 + hi * 8 + q;
                C[(size_t)row * N + col] = acc[mi][ni][q] + bv;
            }
        }
}

// ---------------------------------------------------------------------------
// q = router @ Wq + bq   ([8,512] x [512,512]) — tiny, VALU
// ---------------------------------------------------------------------------
__global__ __launch_bounds__(256) void qproj_kernel(
    const float* __restrict__ router, const float* __restrict__ Wq,
    const float* __restrict__ bq, float* __restrict__ qbuf)
{
    int g = blockIdx.x * 256 + threadIdx.x;       // 4096 = E*D
    int e = g >> 9, d = g & 511;
    float a = bq[d];
    for (int i = 0; i < DIM; ++i)
        a += router[e * DIM + i] * Wq[(size_t)i * DIM + d];
    qbuf[g] = a;
}

// ---------------------------------------------------------------------------
// attention per (batch, head): scores -> softmax -> A@v, E=8 query slots
// ---------------------------------------------------------------------------
__global__ __launch_bounds__(256) void attention_kernel(
    const float* __restrict__ qbuf, const float* __restrict__ kbuf,
    const float* __restrict__ vbuf, float* __restrict__ attb)
{
    __shared__ float qs[NEXP * DK];     // 4 KB
    __shared__ float sc[NEXP * SEQ];    // 16 KB
    const int b = blockIdx.x >> 2;      // H = 4
    const int h = blockIdx.x & 3;
    const int tid = threadIdx.x;

    for (int i = tid; i < NEXP * DK; i += 256) {
        int e = i >> 7, j = i & 127;
        qs[i] = qbuf[e * DIM + h * DK + j];
    }
    __syncthreads();

    const float scale = 0.08838834764831845f;   // 1/sqrt(128)
    for (int p = tid; p < NEXP * SEQ; p += 256) {
        int e = p >> 9, s = p & 511;
        const float* kr = kbuf + ((size_t)(b * SEQ + s)) * DIM + h * DK;
        float d = 0.f;
        #pragma unroll 8
        for (int j = 0; j < DK; ++j) d += qs[e * DK + j] * kr[j];
        sc[e * SEQ + s] = d * scale;
    }
    __syncthreads();

    {   // softmax: one wave per expert row
        const int e = tid >> 5, lane = tid & 31;
        float m = -3.4e38f;
        for (int s = lane; s < SEQ; s += 32) m = fmaxf(m, sc[e * SEQ + s]);
        for (int o = 16; o; o >>= 1) m = fmaxf(m, __shfl_xor(m, o, 32));
        float sum = 0.f;
        for (int s = lane; s < SEQ; s += 32) {
            float ex = __expf(sc[e * SEQ + s] - m);
            sc[e * SEQ + s] = ex;
            sum += ex;
        }
        for (int o = 16; o; o >>= 1) sum += __shfl_xor(sum, o, 32);
        float inv = 1.0f / sum;
        for (int s = lane; s < SEQ; s += 32) sc[e * SEQ + s] *= inv;
    }
    __syncthreads();

    for (int p = tid; p < NEXP * DK; p += 256) {
        int e = p >> 7, j = p & 127;
        const float* vr = vbuf + (size_t)(b * SEQ) * DIM + h * DK + j;
        float a = 0.f;
        for (int s = 0; s < SEQ; ++s) a += sc[e * SEQ + s] * vr[(size_t)s * DIM];
        attb[((size_t)(b * NEXP + e)) * DIM + h * DK + j] = a;
    }
}

// ---------------------------------------------------------------------------
// inv norms of att2 rows ([B*E] rows of length D)
// ---------------------------------------------------------------------------
__global__ __launch_bounds__(128) void norm_kernel(
    const float* __restrict__ att2, float* __restrict__ invna)
{
    __shared__ float red[128];
    const int be = blockIdx.x, tid = threadIdx.x;
    float s = 0.f;
    for (int j = tid; j < DIM; j += 128) {
        float v = att2[(size_t)be * DIM + j];
        s += v * v;
    }
    red[tid] = s;
    __syncthreads();
    for (int o = 64; o; o >>= 1) {
        if (tid < o) red[tid] += red[tid + o];
        __syncthreads();
    }
    if (tid == 0) invna[be] = rsqrtf(red[0]);
}

// ---------------------------------------------------------------------------
// routing: argmax_e (att2[b,e]·x[b,n]) * inv||att2[b,e]||  (monotonic ≡ cosine)
// one wave per token
// ---------------------------------------------------------------------------
__global__ __launch_bounds__(256) void routing_kernel(
    const float* __restrict__ x, const float* __restrict__ att2,
    const float* __restrict__ invna, int* __restrict__ idx)
{
    const int wid = threadIdx.x >> 5, lane = threadIdx.x & 31;
    const int t = blockIdx.x * 8 + wid;
    const int b = t >> 9;
    const float* xr = x + (size_t)t * DIM;
    const float* ar = att2 + (size_t)b * NEXP * DIM;
    float acc[NEXP] = {};
    for (int j = lane; j < DIM; j += 32) {
        float xv = xr[j];
        #pragma unroll
        for (int e = 0; e < NEXP; ++e) acc[e] += xv * ar[e * DIM + j];
    }
    #pragma unroll
    for (int e = 0; e < NEXP; ++e)
        for (int o = 16; o; o >>= 1) acc[e] += __shfl_xor(acc[e], o, 32);
    if (lane == 0) {
        int best = 0;
        float bs = acc[0] * invna[b * NEXP + 0];
        #pragma unroll
        for (int e = 1; e < NEXP; ++e) {
            float s = acc[e] * invna[b * NEXP + e];
            if (s > bs) { bs = s; best = e; }
        }
        idx[t] = best;
    }
}

// ---------------------------------------------------------------------------
// fused per-token: conv1d(2->1, K=25, pad 12) for the SELECTED expert only,
// then Linear(512->96) + gather  (avoids 8x redundant expert work)
// ---------------------------------------------------------------------------
__global__ __launch_bounds__(128) void expert_kernel(
    const float* __restrict__ x, const float* __restrict__ att2,
    const int* __restrict__ idx, const float* __restrict__ Wc,
    const float* __restrict__ bc, const float* __restrict__ Wp,
    const float* __restrict__ bp, float* __restrict__ out)
{
    __shared__ float xz[2 * DIM];
    __shared__ float co[DIM];
    __shared__ float wcs[50];
    const int t = blockIdx.x;
    const int b = t >> 9;
    const int tid = threadIdx.x;
    const int e = idx[t];

    for (int j = tid; j < DIM; j += 128) {
        xz[j]       = x[(size_t)t * DIM + j];
        xz[DIM + j] = att2[((size_t)(b * NEXP + e)) * DIM + j];
    }
    if (tid < 50) wcs[tid] = Wc[e * 50 + tid];
    __syncthreads();

    const float bcv = bc[e];
    for (int l = tid; l < DIM; l += 128) {
        float a = bcv;
        #pragma unroll
        for (int c = 0; c < 2; ++c)
            #pragma unroll
            for (int j = 0; j < 25; ++j) {
                int pos = l - 12 + j;
                if (pos >= 0 && pos < DIM) a += xz[c * DIM + pos] * wcs[c * 25 + j];
            }
        co[l] = a;
    }
    __syncthreads();

    if (tid < POUT) {
        const float* wp = Wp + ((size_t)e * DIM) * POUT + tid;
        float a = bp[e * POUT + tid];
        for (int l = 0; l < DIM; ++l) a += co[l] * wp[(size_t)l * POUT];
        out[(size_t)t * POUT + tid] = a;
    }
}

// ---------------------------------------------------------------------------
extern "C" void kernel_launch(void* const* d_in, const int* in_sizes, int n_in,
                              void* d_out, int out_size, void* d_ws, size_t ws_size,
                              hipStream_t stream) {
    const float* x      = (const float*)d_in[0];
    const float* router = (const float*)d_in[1];
    const float* Wq     = (const float*)d_in[2];
    const float* bq     = (const float*)d_in[3];
    const float* Wk     = (const float*)d_in[4];
    const float* bk     = (const float*)d_in[5];
    const float* Wv     = (const float*)d_in[6];
    const float* bv     = (const float*)d_in[7];
    const float* Wo     = (const float*)d_in[8];
    const float* bo     = (const float*)d_in[9];
    const float* Wc     = (const float*)d_in[10];
    const float* bc     = (const float*)d_in[11];
    const float* Wp     = (const float*)d_in[12];
    const float* bp     = (const float*)d_in[13];
    float* out = (float*)d_out;

    // workspace carve-up (~87 MB total)
    char* ws = (char*)d_ws;
    float* qbuf  = (float*)ws;  ws += (size_t)NEXP * DIM * 4;            // 16 KB
    float* kbuf  = (float*)ws;  ws += (size_t)NTOK * DIM * 4;            // 33.6 MB
    float* vbuf  = (float*)ws;  ws += (size_t)NTOK * DIM * 4;            // 33.6 MB
    float* attb  = (float*)ws;  ws += (size_t)BATCH * NEXP * DIM * 4;    // 512 KB
    float* att2  = (float*)ws;  ws += (size_t)BATCH * NEXP * DIM * 4;    // 512 KB
    float* invna = (float*)ws;  ws += (size_t)BATCH * NEXP * 4;          // 1 KB
    int*   idx   = (int*)ws;    ws += (size_t)NTOK * 4;                  // 64 KB
    u32*   xb    = (u32*)ws;    ws += (size_t)NTOK * (DIM / 2) * 4;      // 16.8 MB (bf16 x)
    u32*   wkt   = (u32*)ws;    ws += (size_t)DIM * (DIM / 2) * 4;       // 512 KB (bf16 Wk^T)
    u32*   wvt   = (u32*)ws;    ws += (size_t)DIM * (DIM / 2) * 4;       // 512 KB
    u32*   wot   = (u32*)ws;    ws += (size_t)DIM * (DIM / 2) * 4;       // 512 KB
    u32*   attbb = (u32*)ws;    ws += (size_t)BATCH * NEXP * (DIM / 2) * 4; // 256 KB

    // pre-pass conversions (one-time, bandwidth-bound)
    cvt_bf16_kernel<<<NTOK * DIM / 2 / 256, 256, 0, stream>>>(x, xb, NTOK * DIM / 2);
    cvt_T_kernel<<<DIM * DIM / 2 / 256, 256, 0, stream>>>(Wk, wkt, DIM, DIM);
    cvt_T_kernel<<<DIM * DIM / 2 / 256, 256, 0, stream>>>(Wv, wvt, DIM, DIM);
    cvt_T_kernel<<<DIM * DIM / 2 / 256, 256, 0, stream>>>(Wo, wot, DIM, DIM);

    qproj_kernel<<<16, 256, 0, stream>>>(router, Wq, bq, qbuf);

    // K / V projections: [16384,512] x [512,512] on the WMMA pipe (TDM-staged)
    gemm_bias_bf16wmma<<<dim3(DIM / BN, NTOK / BM), 256, 0, stream>>>(
        xb, wkt, bk, kbuf, NTOK, DIM, DIM);
    gemm_bias_bf16wmma<<<dim3(DIM / BN, NTOK / BM), 256, 0, stream>>>(
        xb, wvt, bv, vbuf, NTOK, DIM, DIM);

    attention_kernel<<<BATCH * NHEAD, 256, 0, stream>>>(qbuf, kbuf, vbuf, attb);

    // att @ Wo + bo: [256,512] x [512,512]
    cvt_bf16_kernel<<<BATCH * NEXP * DIM / 2 / 256, 256, 0, stream>>>(
        attb, attbb, BATCH * NEXP * DIM / 2);
    gemm_bias_bf16wmma<<<dim3(DIM / BN, (BATCH * NEXP) / BM), 256, 0, stream>>>(
        attbb, wot, bo, att2, BATCH * NEXP, DIM, DIM);

    norm_kernel<<<BATCH * NEXP, 128, 0, stream>>>(att2, invna);
    routing_kernel<<<NTOK / 8, 256, 0, stream>>>(x, att2, invna, idx);
    expert_kernel<<<NTOK, 128, 0, stream>>>(x, att2, idx, Wc, bc, Wp, bp, out);
}
